// SeqMoELogits_88596585381972
// MI455X (gfx1250) — matrix operand order repaired and checked by reading, not compiled, not run on hardware
//
#include <hip/hip_runtime.h>
#include <hip/hip_bf16.h>

// Problem constants (match reference)
#define B_ 8192
#define D_ 1024
#define E_ 8
#define H_ 2048
#define C_ 512
#define LN_EPS 1e-5f

typedef __attribute__((ext_vector_type(16))) _Float16 v16h;
typedef __attribute__((ext_vector_type(8)))  _Float16 v8h;
typedef __attribute__((ext_vector_type(8)))  float    v8f;

static constexpr int MT = 32;    // token tile (rows): 2 M-tiles share every B fragment
static constexpr int HB = 128;   // H chunk kept in LDS

__device__ inline float wave_reduce_add(float v) {
#pragma unroll
  for (int off = 16; off > 0; off >>= 1) v += __shfl_xor(v, off, 32);
  return v;
}

// A-fragment loader (f16 16x16x32): lane (m=L%16, half=L/16) holds halves
// K = [8h..8h+7] and [16+8h..16+8h+7] -> two aligned 16B chunks.
__device__ inline v16h load_a_frag(const _Float16* rowp, int off) {
  const v8h lo = *reinterpret_cast<const v8h*>(rowp + off);
  const v8h hi = *reinterpret_cast<const v8h*>(rowp + off + 16);
  v16h a;
#pragma unroll
  for (int j = 0; j < 8; ++j) { a[j] = lo[j]; a[j + 8] = hi[j]; }
  return a;
}

// ---------------------------------------------------------------------------
// Kernel 1: LayerNorm (stats) + router top-2 softmax + token scatter.
// One wave (32 lanes) per token; 8 waves per block.
// ---------------------------------------------------------------------------
__global__ __launch_bounds__(256)
void router_ln_kernel(const float* __restrict__ x,
                      const float* __restrict__ Wr,
                      const float* __restrict__ br,
                      _Float16* __restrict__ xn,
                      int* __restrict__ cnt,
                      int* __restrict__ tokidx,
                      float* __restrict__ tokw) {
  const int wave = threadIdx.x >> 5;
  const int lane = threadIdx.x & 31;
  const int b = blockIdx.x * 8 + wave;   // B_ is a multiple of 8

  float xv[32];
  const float* xp = x + (size_t)b * D_;
#pragma unroll
  for (int i = 0; i < 32; ++i) xv[i] = xp[lane + 32 * i];

  // mean
  float s = 0.f;
#pragma unroll
  for (int i = 0; i < 32; ++i) s += xv[i];
  s = wave_reduce_add(s);
  const float mu = s * (1.0f / D_);

  // variance (two-pass, matches reference)
  float vs = 0.f;
#pragma unroll
  for (int i = 0; i < 32; ++i) { float d = xv[i] - mu; vs += d * d; }
  vs = wave_reduce_add(vs);
  const float rsig = rsqrtf(vs * (1.0f / D_) + LN_EPS);

  // write normalized activations as f16 (gamma folded into W1h, beta into c1)
  _Float16* xo = xn + (size_t)b * D_;
#pragma unroll
  for (int i = 0; i < 32; ++i)
    xo[lane + 32 * i] = (_Float16)((xv[i] - mu) * rsig);

  // router logits on raw x (as in reference)
  float lg[E_];
#pragma unroll
  for (int e = 0; e < E_; ++e) {
    const float* wp = Wr + (size_t)e * D_;
    float t = 0.f;
#pragma unroll
    for (int i = 0; i < 32; ++i) t += xv[i] * wp[lane + 32 * i];
    lg[e] = wave_reduce_add(t) + br[e];
  }

  // top-2 (strict > keeps earliest index on ties, like jax.lax.top_k)
  int i0 = 0; float v0 = lg[0];
#pragma unroll
  for (int e = 1; e < E_; ++e) if (lg[e] > v0) { v0 = lg[e]; i0 = e; }
  int i1 = -1; float v1 = -3.402823466e38f;
#pragma unroll
  for (int e = 0; e < E_; ++e) if (e != i0 && lg[e] > v1) { v1 = lg[e]; i1 = e; }

  // softmax over the two selected logits (stable)
  const float ex = __expf(v1 - v0);
  const float g0 = 1.0f / (1.0f + ex);
  const float g1 = ex * g0;

  if (lane == 0) {
    int p0 = atomicAdd(&cnt[i0], 1);
    tokidx[i0 * B_ + p0] = b; tokw[i0 * B_ + p0] = g0;
    int p1 = atomicAdd(&cnt[i1], 1);
    tokidx[i1 * B_ + p1] = b; tokw[i1 * B_ + p1] = g1;
  }
}

// ---------------------------------------------------------------------------
// Kernel 2a: fold LN shift into FC1 bias: c1[e,h] = b1[e,h] + sum_d beta[e,d]*W1[e,h,d]
// ---------------------------------------------------------------------------
__global__ __launch_bounds__(256)
void c1_fold_kernel(const float* __restrict__ beta,
                    const float* __restrict__ W1,
                    const float* __restrict__ b1,
                    float* __restrict__ c1) {
  const int wave = threadIdx.x >> 5;
  const int lane = threadIdx.x & 31;
  const int idx = blockIdx.x * 8 + wave;        // 0 .. E*H-1
  const int e = idx / H_;
  const float* bp = beta + (size_t)e * D_;
  const float* wp = W1 + (size_t)idx * D_;
  float s = 0.f;
  for (int i = lane; i < D_; i += 32) s += bp[i] * wp[i];
  s = wave_reduce_add(s);
  if (lane == 0) c1[idx] = s + b1[idx];
}

// ---------------------------------------------------------------------------
// Kernel 2b: W1h[e,h,d] = f16(W1[e,h,d] * gamma[e,d])  (one-time conversion)
// ---------------------------------------------------------------------------
__global__ __launch_bounds__(256)
void w1_prep_kernel(const float* __restrict__ W1,
                    const float* __restrict__ gamma,
                    _Float16* __restrict__ W1h) {
  const size_t i = (size_t)blockIdx.x * 256 + threadIdx.x;   // grid covers E*H*D/4
  const size_t n0 = i * 4;
  const size_t e = n0 / ((size_t)H_ * D_);
  const int d = (int)(n0 & (D_ - 1));
  const float4 w = *reinterpret_cast<const float4*>(W1 + n0);
  const float4 g = *reinterpret_cast<const float4*>(gamma + e * D_ + d);
  _Float16* o = W1h + n0;
  o[0] = (_Float16)(w.x * g.x); o[1] = (_Float16)(w.y * g.y);
  o[2] = (_Float16)(w.z * g.z); o[3] = (_Float16)(w.w * g.w);
}

// ---------------------------------------------------------------------------
// Kernel 2c: W2h = f16(W2)
// ---------------------------------------------------------------------------
__global__ __launch_bounds__(256)
void w2_prep_kernel(const float* __restrict__ W2, _Float16* __restrict__ W2h) {
  const size_t i = (size_t)blockIdx.x * 256 + threadIdx.x;   // grid covers E*C*H/4
  const size_t n0 = i * 4;
  const float4 w = *reinterpret_cast<const float4*>(W2 + n0);
  _Float16* o = W2h + n0;
  o[0] = (_Float16)w.x; o[1] = (_Float16)w.y;
  o[2] = (_Float16)w.z; o[3] = (_Float16)w.w;
}

// ---------------------------------------------------------------------------
// Kernel 3: fused per-expert FFN over gathered 32-token tiles.
//   grid = (B/MT, E). Block = 256 threads = 8 waves.
//   FC1: h = gelu(xn @ W1h[e]^T + c1[e])  (H chunked by HB, h in LDS)
//   FC2: y += h_chunk @ W2h[e]^T ; epilogue: out[b] += gate * (y + b2[e])
// Two M-tiles (rows 0-15 / 16-31) share every B fragment: 2 WMMAs per
// 32B weight load, halving per-FLOP L2 traffic vs MT=16.
// ---------------------------------------------------------------------------
__global__ __launch_bounds__(256)
void moe_ffn_kernel(const _Float16* __restrict__ xn,
                    const _Float16* __restrict__ W1h,
                    const float* __restrict__ c1,
                    const _Float16* __restrict__ W2h,
                    const float* __restrict__ b2,
                    const int* __restrict__ cnt,
                    const int* __restrict__ tokidx,
                    const float* __restrict__ tokw,
                    float* __restrict__ out) {
  const int e = blockIdx.y;
  const int n = cnt[e];
  const int row0 = blockIdx.x * MT;
  if (row0 >= n) return;   // block-uniform early exit (sparse routing: ~4x fewer tiles)

  // Row strides (D_+16)*2 = 2080 B and (HB+8)*2 = 272 B are 16B multiples and
  // skew LDS banks between rows; all fragment chunks below are 16B-aligned.
  __shared__ __attribute__((aligned(32))) _Float16 Xs[MT][D_ + 16];
  __shared__ __attribute__((aligned(32))) _Float16 Hs[MT][HB + 8];
  __shared__ int   gidx[MT];
  __shared__ float gw[MT];

  const int tid = threadIdx.x;
  const int wave = tid >> 5;
  const int lane = tid & 31;
  const int half = lane >> 4;
  const int l16 = lane & 15;

  if (tid < MT) {
    int p = row0 + tid;
    if (p < n) { gidx[tid] = tokidx[e * B_ + p]; gw[tid] = tokw[e * B_ + p]; }
    else       { gidx[tid] = -1;                 gw[tid] = 0.f; }
  }
  __syncthreads();

  // Stage Xs (pure gather copy, 8 halves per thread-iteration, coalesced)
  for (int idx = tid; idx < MT * (D_ / 8); idx += 256) {
    const int r = idx >> 7;              // D_/8 = 128
    const int d8 = (idx & 127) * 8;
    const int b = gidx[r];
    v8h v = {};
    if (b >= 0) v = *reinterpret_cast<const v8h*>(xn + (size_t)b * D_ + d8);
    *reinterpret_cast<v8h*>(&Xs[r][d8]) = v;
  }
  __syncthreads();

  // FC2 accumulators: each wave owns 64 cols x 32 rows (2 M-tiles x 4 N-tiles)
  v8f acc2[2][4] = {};

  for (int hb = 0; hb < H_; hb += HB) {
    const int hcol = hb + wave * 16 + l16;       // this lane's FC1 output column
    const _Float16* w1row = W1h + ((size_t)(e * H_ + hcol)) * D_;

    // Prefetch this chunk's FC2 weight rows into cache while FC1 runs
    // (lowers to global_prefetch_b8; one line per (nt) tile row).
#pragma unroll
    for (int nt = 0; nt < 4; ++nt) {
      const int c = wave * 64 + nt * 16 + l16;
      __builtin_prefetch(W2h + ((size_t)(e * C_ + c)) * H_ + hb + 16 * half, 0, 1);
    }

    // ---------------- FC1 chunk: [MT x HB], each wave owns 16 cols ----------
    v8f acc1[2] = {};
    for (int d0 = 0; d0 < D_; d0 += 32) {
      const v16h a0 = load_a_frag(&Xs[l16][0],      d0 + 8 * half);
      const v16h a1 = load_a_frag(&Xs[16 + l16][0], d0 + 8 * half);
      const v16h bf = load_a_frag(w1row, d0 + 8 * half);
      acc1[0] = __builtin_amdgcn_wmma_f32_16x16x32_f16(false, a0, false, bf,
                                                       (short)0, acc1[0], false, false);
      acc1[1] = __builtin_amdgcn_wmma_f32_16x16x32_f16(false, a1, false, bf,
                                                       (short)0, acc1[1], false, false);
    }

    __syncthreads();   // previous FC2 reads of Hs complete before overwrite

    // bias + exact GELU -> Hs (f16)
    {
      const float cb = c1[e * H_ + hcol];
      const int ncol = wave * 16 + l16;
#pragma unroll
      for (int mt = 0; mt < 2; ++mt) {
#pragma unroll
        for (int r = 0; r < 8; ++r) {
          const int m = mt * 16 + r + 8 * half;
          float v = acc1[mt][r] + cb;
          v = 0.5f * v * (1.0f + erff(v * 0.70710678118654752f));
          Hs[m][ncol] = (_Float16)v;
        }
      }
    }
    __syncthreads();

    // ---------------- FC2 partial over this H chunk ------------------------
    for (int kk = 0; kk < HB; kk += 32) {
      const v16h ah0 = load_a_frag(&Hs[l16][0],      kk + 8 * half);
      const v16h ah1 = load_a_frag(&Hs[16 + l16][0], kk + 8 * half);
#pragma unroll
      for (int nt = 0; nt < 4; ++nt) {
        const int c = wave * 64 + nt * 16 + l16;
        const v16h bf = load_a_frag(
            W2h + ((size_t)(e * C_ + c)) * H_, hb + kk + 8 * half);
        acc2[0][nt] = __builtin_amdgcn_wmma_f32_16x16x32_f16(false, ah0, false, bf,
                                                             (short)0, acc2[0][nt], false, false);
        acc2[1][nt] = __builtin_amdgcn_wmma_f32_16x16x32_f16(false, ah1, false, bf,
                                                             (short)0, acc2[1][nt], false, false);
      }
    }
  }

  // Epilogue: gate-weighted scatter-add. Exactly K=2 commutative float adds
  // per out element (zero-initialized) -> bitwise deterministic.
#pragma unroll
  for (int nt = 0; nt < 4; ++nt) {
    const int c = wave * 64 + nt * 16 + l16;
    const float bb = b2[e * C_ + c];
#pragma unroll
    for (int mt = 0; mt < 2; ++mt) {
#pragma unroll
      for (int r = 0; r < 8; ++r) {
        const int m = mt * 16 + r + 8 * half;
        const int b = gidx[m];
        if (b >= 0) {
          const float val = (acc2[mt][nt][r] + bb) * gw[m];
          atomicAdd(&out[(size_t)b * C_ + c], val);
        }
      }
    }
  }
}

// ---------------------------------------------------------------------------
extern "C" void kernel_launch(void* const* d_in, const int* in_sizes, int n_in,
                              void* d_out, int out_size, void* d_ws, size_t ws_size,
                              hipStream_t stream) {
  (void)in_sizes; (void)n_in; (void)out_size; (void)ws_size;

  const float* x     = (const float*)d_in[0];
  const float* Wr    = (const float*)d_in[1];
  const float* br    = (const float*)d_in[2];
  const float* gamma = (const float*)d_in[3];
  const float* beta  = (const float*)d_in[4];
  const float* W1    = (const float*)d_in[5];
  const float* b1    = (const float*)d_in[6];
  const float* W2    = (const float*)d_in[7];
  const float* b2    = (const float*)d_in[8];
  float* out = (float*)d_out;

  // Workspace layout (~65 MB total, all offsets 256B-aligned)
  char* ws = (char*)d_ws;
  size_t off = 0;
  int*      cnt    = (int*)(ws + off);        off += 256;
  int*      tokidx = (int*)(ws + off);        off += (size_t)E_ * B_ * 4;
  float*    tokw   = (float*)(ws + off);      off += (size_t)E_ * B_ * 4;
  float*    c1     = (float*)(ws + off);      off += (size_t)E_ * H_ * 4;
  _Float16* xn     = (_Float16*)(ws + off);   off += (size_t)B_ * D_ * 2;
  _Float16* W1h    = (_Float16*)(ws + off);   off += (size_t)E_ * H_ * D_ * 2;
  _Float16* W2h    = (_Float16*)(ws + off);   off += (size_t)E_ * C_ * H_ * 2;

  hipMemsetAsync(out, 0, (size_t)B_ * C_ * sizeof(float), stream);
  hipMemsetAsync(cnt, 0, 256, stream);

  router_ln_kernel<<<B_ / 8, 256, 0, stream>>>(x, Wr, br, xn, cnt, tokidx, tokw);
  c1_fold_kernel<<<(E_ * H_) / 8, 256, 0, stream>>>(beta, W1, b1, c1);
  w1_prep_kernel<<<(E_ * H_ * D_ / 4) / 256, 256, 0, stream>>>(W1, gamma, W1h);
  w2_prep_kernel<<<(E_ * C_ * H_ / 4) / 256, 256, 0, stream>>>(W2, W2h);
  moe_ffn_kernel<<<dim3(B_ / MT, E_), 256, 0, stream>>>(xn, W1h, c1, W2h, b2,
                                                        cnt, tokidx, tokw, out);
}
